// WeightOnlyQLinear_33182917329676
// MI455X (gfx1250) — matrix operand order repaired and checked by reading, not compiled, not run on hardware
//
#include <hip/hip_runtime.h>

typedef __attribute__((ext_vector_type(16))) _Float16 v16h;
typedef __attribute__((ext_vector_type(8)))  _Float16 v8h;
typedef __attribute__((ext_vector_type(8)))  float    v8f;
typedef __attribute__((ext_vector_type(4)))  int      v4i;

// Problem constants (fixed by the reference)
constexpr int Mdim  = 8192;
constexpr int Kdim  = 4096;
constexpr int Ndim  = 11008;
constexpr int GROUP = 128;
constexpr int NWRD  = Ndim / 8;   // packed int32 words per K-row = 1376

// Tiling: 256x128x64 block, 8 wave32 waves in 4x2, each wave owns 64x64
constexpr int BM = 256, BN = 128, BK = 64;
constexpr int WM = 64,  WN = 64;
constexpr int LDA = BK + 8;       // 72 halves (144B): bank-spread, 16B-aligned
constexpr int LDB = BK + 8;
constexpr int KTILES = Kdim / BK; // 64

// ---- CDNA5 async global->LDS copy (ASYNCcnt path), 16B per lane ----
__device__ __forceinline__ void async_copy_b128(const _Float16* gsrc, _Float16* ldst) {
#if __has_builtin(__builtin_amdgcn_global_load_async_to_lds_b128)
    __builtin_amdgcn_global_load_async_to_lds_b128(
        (__attribute__((address_space(1))) v4i*)gsrc,
        (__attribute__((address_space(3))) v4i*)ldst,
        /*offset=*/0, /*cpol=*/0);
#else
    // VDST = per-lane LDS byte offset (flat addr low 32 bits per LDS aperture),
    // VADDR = 64-bit global address pair.
    unsigned int loff = (unsigned int)(unsigned long long)ldst;
    asm volatile("global_load_async_to_lds_b128 %0, %1, off"
                 :: "v"(loff), "v"(gsrc) : "memory");
#endif
}

__device__ __forceinline__ void wait_asynccnt0() {
#if __has_builtin(__builtin_amdgcn_s_wait_asynccnt)
    __builtin_amdgcn_s_wait_asynccnt(0);
#else
    asm volatile("s_wait_asynccnt 0x0" ::: "memory");
#endif
}

__global__ __launch_bounds__(256)
void awq_gemm_wmma_f16(const _Float16* __restrict__ x,
                       const int*      __restrict__ qweight,
                       const _Float16* __restrict__ scales,
                       const int*      __restrict__ qzeros,
                       const _Float16* __restrict__ bias,
                       _Float16*       __restrict__ out)
{
    // Ping-pong buffers: 2 x (36KB A + 18KB B) = 108KB of the WGP's 320KB LDS
    __shared__ __align__(64) _Float16 As[2][BM * LDA];   // row-major [BM][LDA]
    __shared__ __align__(64) _Float16 Bs[2][BN * LDB];   // n-major  [BN][LDB]

    const int tid    = threadIdx.x;
    const int wave   = tid >> 5;
    const int lane   = tid & 31;
    const int lane16 = lane & 15;     // row (A frag) / column (B frag) within 16
    const int half   = lane >> 4;     // K-half selector per ISA 16-bit layouts

    const int wave_m = wave >> 1;     // 0..3
    const int wave_n = wave & 1;      // 0..1

    const int m0 = blockIdx.y * BM;
    const int n0 = blockIdx.x * BN;

    v8f acc[4][4] = {};               // 4 M-tiles x 4 N-tiles of 16x16 f32

    // ---------------- prologue: fill buffer 0 with tile 0 ----------------
    {
        const int k0 = 0;
        #pragma unroll
        for (int i = 0; i < 8; ++i) {              // A: 256x64 halves, async DMA to LDS
            const int c   = tid + i * 256;
            const int row = c >> 3;
            const int seg = c & 7;
            async_copy_b128(x + (size_t)(m0 + row) * Kdim + k0 + seg * 8,
                            As[0] + row * LDA + seg * 8);
        }
        const int g = 0;
        #pragma unroll
        for (int i = 0; i < 4; ++i) {              // B: 64 k-rows x 16 words, dequant
            const int w  = tid + i * 256;
            const int kl = w >> 4;
            const int wc = w & 15;
            const int qw = qweight[(size_t)(k0 + kl) * NWRD + (n0 >> 3) + wc];
            const int zw = qzeros [(size_t)g * NWRD + (n0 >> 3) + wc];
            const _Float16* sc = scales + (size_t)g * Ndim + n0 + wc * 8;
            _Float16* bdst = Bs[0] + (wc * 8) * LDB + kl;
            #pragma unroll
            for (int nib = 0; nib < 8; ++nib) {
                const int colo = ((2 * nib) & 7) | (nib >> 2);  // AWQ nibble order
                const int wn = (qw >> (4 * nib)) & 0xF;
                const int zn = (zw >> (4 * nib)) & 0xF;
                bdst[colo * LDB] = (_Float16)(wn - zn) * sc[colo];
            }
        }
        wait_asynccnt0();
    }
    __syncthreads();

    // ---------------- pipelined main loop ----------------
    for (int t = 0; t < KTILES; ++t) {
        const int cur = t & 1;
        const int nxt = cur ^ 1;
        const bool has_next = (t + 1) < KTILES;

        // ---- kick off next tile: A goes straight to LDS via async DMA (no VGPR
        //      staging); qweight/zeros/scales staged in a few registers.
        //      Writes to buf[nxt] are safe: barrier ending t-1 retired its readers.
        int qw[4], zw[4];
        v8h screg[4];
        if (has_next) {
            const int k0n = (t + 1) * BK;
            const int gn  = k0n / GROUP;       // BK=64 aligned -> single group per tile
            #pragma unroll
            for (int i = 0; i < 8; ++i) {
                const int c   = tid + i * 256;
                const int row = c >> 3;
                const int seg = c & 7;
                async_copy_b128(x + (size_t)(m0 + row) * Kdim + k0n + seg * 8,
                                As[nxt] + row * LDA + seg * 8);
            }
            #pragma unroll
            for (int i = 0; i < 4; ++i) {
                const int w  = tid + i * 256;
                const int kl = w >> 4;
                const int wc = w & 15;
                qw[i] = qweight[(size_t)(k0n + kl) * NWRD + (n0 >> 3) + wc];
                zw[i] = qzeros [(size_t)gn * NWRD + (n0 >> 3) + wc];
                screg[i] = *(const v8h*)(scales + (size_t)gn * Ndim + n0 + wc * 8);
            }
            if (t + 2 < KTILES) {  // gfx1250 speculative prefetch, two tiles ahead
                __builtin_prefetch(x + (size_t)(m0 + (tid >> 1)) * Kdim + k0n + BK, 0, 0);
                __builtin_prefetch(qweight + (size_t)(k0n + BK + (tid & 63)) * NWRD + (n0 >> 3), 0, 0);
            }
        }

        // ---- compute on current buffer: 2 sub-steps of K=32, 32 WMMAs
        #pragma unroll
        for (int kk = 0; kk < BK; kk += 32) {
            // A frag (16-bit A 16x32): lanes0-15 K 0..7|16..23, lanes16-31 K 8..15|24..31
            v16h afrag[4];
            #pragma unroll
            for (int mi = 0; mi < 4; ++mi) {
                const _Float16* Ar =
                    As[cur] + (wave_m * WM + mi * 16 + lane16) * LDA + kk + half * 8;
                v8h lo = *(const v8h*)(Ar);
                v8h hi = *(const v8h*)(Ar + 16);
                afrag[mi] = __builtin_shufflevector(
                    lo, hi, 0,1,2,3,4,5,6,7,8,9,10,11,12,13,14,15);
            }
            // B frag (16-bit B 32x16): lanes0-15 K 0..7|16..23, lanes16-31 K 8..15|24..31
            v16h bfrag[4];
            #pragma unroll
            for (int ni = 0; ni < 4; ++ni) {
                const _Float16* Br =
                    Bs[cur] + (wave_n * WN + ni * 16 + lane16) * LDB + kk + half * 8;
                v8h lo = *(const v8h*)(Br);
                v8h hi = *(const v8h*)(Br + 16);
                bfrag[ni] = __builtin_shufflevector(
                    lo, hi, 0,1,2,3,4,5,6,7,8,9,10,11,12,13,14,15);
            }
            #pragma unroll
            for (int mi = 0; mi < 4; ++mi)
                #pragma unroll
                for (int ni = 0; ni < 4; ++ni)
                    acc[mi][ni] = __builtin_amdgcn_wmma_f32_16x16x32_f16(
                        false, afrag[mi], false, bfrag[ni],
                        (short)0, acc[mi][ni], false, false);
        }

        // ---- dequant staged B words into the next buffer
        if (has_next) {
            #pragma unroll
            for (int i = 0; i < 4; ++i) {
                const int w  = tid + i * 256;
                const int kl = w >> 4;
                const int wc = w & 15;
                _Float16* bdst = Bs[nxt] + (wc * 8) * LDB + kl;
                #pragma unroll
                for (int nib = 0; nib < 8; ++nib) {
                    const int colo = ((2 * nib) & 7) | (nib >> 2);
                    const int wn = (qw[i] >> (4 * nib)) & 0xF;
                    const int zn = (zw[i] >> (4 * nib)) & 0xF;
                    bdst[colo * LDB] = (_Float16)(wn - zn) * screg[i][colo];
                }
            }
            wait_asynccnt0();   // A-tile DMA for t+1 must be in LDS before the barrier
        }
        __syncthreads();
    }

    // ---------------- epilogue ----------------
    // C/D 16x16 f32 layout: VGPR r -> (M = r + 8*half, N = lane16)
    #pragma unroll
    for (int ni = 0; ni < 4; ++ni) {
        const int col = n0 + wave_n * WN + ni * 16 + lane16;
        const float bf = (float)bias[col];
        #pragma unroll
        for (int mi = 0; mi < 4; ++mi) {
            const int rbase = m0 + wave_m * WM + mi * 16 + half * 8;
            #pragma unroll
            for (int r = 0; r < 8; ++r) {
                out[(size_t)(rbase + r) * Ndim + col] = (_Float16)(acc[mi][ni][r] + bf);
            }
        }
    }
}

extern "C" void kernel_launch(void* const* d_in, const int* in_sizes, int n_in,
                              void* d_out, int out_size, void* d_ws, size_t ws_size,
                              hipStream_t stream) {
    (void)in_sizes; (void)n_in; (void)out_size; (void)d_ws; (void)ws_size;
    const _Float16* x       = (const _Float16*)d_in[0];
    const int*      qweight = (const int*)     d_in[1];
    const _Float16* scales  = (const _Float16*)d_in[2];
    const int*      qzeros  = (const int*)     d_in[3];
    const _Float16* bias    = (const _Float16*)d_in[4];
    // d_in[5] = group_size scalar (hardcoded GROUP=128)
    _Float16* out = (_Float16*)d_out;

    dim3 grid(Ndim / BN, Mdim / BM);   // (86, 32)
    dim3 block(256);
    awq_gemm_wmma_f16<<<grid, block, 0, stream>>>(x, qweight, scales, qzeros, bias, out);
}
